// CrossAxisAttention_23991687316085
// MI455X (gfx1250) — matrix operand order repaired and checked by reading, not compiled, not run on hardware
//
#include <hip/hip_runtime.h>
#include <stdint.h>

// ---------------------------------------------------------------------------
// CrossAxisAttention for MI455X (gfx1250, wave32, WMMA, TDM)
// LN -> depthwise(7/11/21) x/y -> WMMA GEMM (w_in) -> pack to_h/to_w
// -> L2 row norms -> WMMA attention (QK^T via LDS-staged K, softmax,
//    AV via LDS-staged V, +Qn) -> out_x+out_y -> WMMA GEMM (w_out)+2b+x
// All LDS staging is double-buffered (stage tile i+1 while WMMA eats tile i).
// ---------------------------------------------------------------------------

#define B_   16
#define C_   256
#define H_   80
#define W_   80
#define HW_  6400
#define HEADS_ 8
#define CP_  32
#define D_   2560           // W_*CP_ == H_*CP_
#define NPIX_ (B_*HW_)      // 102400
#define SZ_  ((size_t)B_*C_*H_*W_)   // 26214400 elements

typedef __attribute__((ext_vector_type(16))) __bf16 v16bf;
typedef __attribute__((ext_vector_type(8)))  float  v8f;
typedef unsigned uv4 __attribute__((ext_vector_type(4)));
typedef int      iv8 __attribute__((ext_vector_type(8)));
typedef int      iv4 __attribute__((ext_vector_type(4)));

union AF  { v16bf v; unsigned u[8]; };
union Pk8 { __bf16 h[8]; uint4 q; };

__device__ __forceinline__ v8f vzero8() {
  v8f z;
#pragma unroll
  for (int i = 0; i < 8; ++i) z[i] = 0.0f;
  return z;
}

__device__ __forceinline__ v8f wmma_bf16(v16bf a, v16bf b, v8f c) {
  // D = A(16x32 bf16) * B(32x16 bf16) + C(16x16 f32)
  return __builtin_amdgcn_wmma_f32_16x16x32_bf16(false, a, false, b,
                                                 (short)0, c, false, false);
}

// A-fragment K index for element e, lane-half hf (16-bit A 16x32 layout)
__device__ __forceinline__ int amapK(int e, int hf) {
  return (e < 8) ? (e + 8 * hf) : (e + 8 + 8 * hf);
}

#if defined(__has_builtin)
#if __has_builtin(__builtin_amdgcn_tensor_load_to_lds) && \
    __has_builtin(__builtin_amdgcn_s_wait_tensorcnt)
#define USE_TDM 1
#endif
#endif

#if defined(USE_TDM)
// Issue a TDM 2D tile load: 128 rows x 32 bf16 cols from a row-major
// [rows x 256] bf16 matrix into LDS (row-major, stride 32 elements).
__device__ __forceinline__ void tdm_issue_A(const __bf16* gsrc, __bf16* lds_dst) {
  unsigned long long ga = (unsigned long long)(const void*)gsrc;
  unsigned lds_off = (unsigned)(unsigned long long)
      (__attribute__((address_space(3))) const void*)lds_dst;
  uv4 g0;
  g0[0] = 1u;                                   // count=1, normal mode
  g0[1] = lds_off;                              // lds_addr (bytes)
  g0[2] = (unsigned)(ga & 0xffffffffu);         // global_addr[31:0]
  g0[3] = (unsigned)(ga >> 32) | 0x80000000u;   // global_addr[56:32] | type=2
  iv8 g1;
  g1[0] = 1 << 16;                              // data_size=2B, mask=0
  g1[1] = (int)(256u << 16);                    // tensor_dim0 = 256 (low16)
  g1[2] = (int)(256u << 16);                    // dim0 hi=0 | tensor_dim1 lo16=256
  g1[3] = (int)(32u << 16);                     // dim1 hi=0 | tile_dim0 = 32
  g1[4] = 128;                                  // tile_dim1 = 128, tile_dim2=0
  g1[5] = 256;                                  // tensor_dim0_stride = 256
  g1[6] = 0; g1[7] = 0;                         // dim1_stride unused (2D)
  iv4 z4; z4[0] = 0; z4[1] = 0; z4[2] = 0; z4[3] = 0;
  iv8 z8; z8[0] = 0; z8[1] = 0; z8[2] = 0; z8[3] = 0;
  z8[4] = 0; z8[5] = 0; z8[6] = 0; z8[7] = 0;
  __builtin_amdgcn_tensor_load_to_lds(g0, g1, z4, z4, z8, 0);
}
#endif

// ---------------------------------------------------------------------------
// 0. weight conversion fp32 -> bf16
// ---------------------------------------------------------------------------
__global__ void prep_weights_kernel(const float* __restrict__ w_in,
                                    const float* __restrict__ w_out,
                                    __bf16* __restrict__ w_in_bf,
                                    __bf16* __restrict__ w_out_bf) {
  int i = blockIdx.x * blockDim.x + threadIdx.x;
  if (i < C_ * C_) w_in_bf[i] = (__bf16)w_in[i];
  else {
    int j = i - C_ * C_;
    if (j < C_ * C_) w_out_bf[j] = (__bf16)w_out[j];
  }
}

// ---------------------------------------------------------------------------
// 1. LayerNorm over channels (per pixel), fp32 -> bf16
// ---------------------------------------------------------------------------
__global__ void ln_kernel(const float* __restrict__ x,
                          const float* __restrict__ g,
                          const float* __restrict__ b,
                          __bf16* __restrict__ xn) {
  int pg = blockIdx.x * blockDim.x + threadIdx.x;   // 0..B*HW-1
  if (pg >= NPIX_) return;
  int bb = pg / HW_;
  int p  = pg % HW_;
  const float* xb = x + (size_t)bb * C_ * HW_ + p;
  float s = 0.f, s2 = 0.f;
#pragma unroll 4
  for (int c = 0; c < C_; ++c) {
    float v = xb[(size_t)c * HW_];
    s += v; s2 += v * v;
  }
  float mu  = s * (1.0f / C_);
  float var = s2 * (1.0f / C_) - mu * mu;
  float inv = rsqrtf(var + 1e-5f);
  __bf16* ob = xn + (size_t)bb * C_ * HW_ + p;
#pragma unroll 4
  for (int c = 0; c < C_; ++c) {
    float v = (xb[(size_t)c * HW_] - mu) * inv * g[c] + b[c];
    ob[(size_t)c * HW_] = (__bf16)v;
  }
}

// ---------------------------------------------------------------------------
// 2a. depthwise along W: sum of k=7,11,21 convs + biases
// ---------------------------------------------------------------------------
__global__ void dwx_kernel(const __bf16* __restrict__ xn,
                           const float* __restrict__ w7,  const float* __restrict__ b7,
                           const float* __restrict__ w11, const float* __restrict__ b11,
                           const float* __restrict__ w21, const float* __restrict__ b21,
                           __bf16* __restrict__ sx) {
  int rid = blockIdx.x;                 // (b*C + c)*H + h
  size_t base = (size_t)rid * W_;
  int c = (rid / H_) % C_;
  __shared__ float row[W_ + 20];
  int t = threadIdx.x;
  if (t < W_ + 20) {
    int w = t - 10;
    row[t] = (w >= 0 && w < W_) ? (float)xn[base + w] : 0.0f;
  }
  __syncthreads();
  if (t < W_) {
    float acc = b7[c] + b11[c] + b21[c];
    const float* k7  = w7  + c * 7;
    const float* k11 = w11 + c * 11;
    const float* k21 = w21 + c * 21;
#pragma unroll
    for (int j = 0; j < 7;  ++j) acc += row[t + 10 + j - 3]  * k7[j];
#pragma unroll
    for (int j = 0; j < 11; ++j) acc += row[t + 10 + j - 5]  * k11[j];
#pragma unroll
    for (int j = 0; j < 21; ++j) acc += row[t + 10 + j - 10] * k21[j];
    sx[base + t] = (__bf16)acc;
  }
}

// ---------------------------------------------------------------------------
// 2b. depthwise along H
// ---------------------------------------------------------------------------
__global__ void dwy_kernel(const __bf16* __restrict__ xn,
                           const float* __restrict__ w7,  const float* __restrict__ b7,
                           const float* __restrict__ w11, const float* __restrict__ b11,
                           const float* __restrict__ w21, const float* __restrict__ b21,
                           __bf16* __restrict__ sy) {
  int bc = blockIdx.x;                  // b*C + c
  int h0 = blockIdx.y * 16;
  int c  = bc & (C_ - 1);
  size_t base = (size_t)bc * HW_;
  __shared__ float tile[36][W_];
  for (int idx = threadIdx.x; idx < 36 * W_; idx += blockDim.x) {
    int r = idx / W_, w = idx % W_;
    int h = h0 - 10 + r;
    tile[r][w] = (h >= 0 && h < H_) ? (float)xn[base + (size_t)h * W_ + w] : 0.0f;
  }
  __syncthreads();
  const float* k7  = w7  + c * 7;
  const float* k11 = w11 + c * 11;
  const float* k21 = w21 + c * 21;
  float bias = b7[c] + b11[c] + b21[c];
  for (int idx = threadIdx.x; idx < 16 * W_; idx += blockDim.x) {
    int hh = idx / W_, w = idx % W_;
    int rr = hh + 10;
    float acc = bias;
#pragma unroll
    for (int j = 0; j < 7;  ++j) acc += tile[rr + j - 3][w]  * k7[j];
#pragma unroll
    for (int j = 0; j < 11; ++j) acc += tile[rr + j - 5][w]  * k11[j];
#pragma unroll
    for (int j = 0; j < 21; ++j) acc += tile[rr + j - 10][w] * k21[j];
    sy[base + (size_t)(h0 + hh) * W_ + w] = (__bf16)acc;
  }
}

// ---------------------------------------------------------------------------
// 3. WMMA GEMM: out[b,m,n] = sum_k A[m,k]*X[b,k,n] (+bias[,resid])
//    block: 256 threads (8 waves), tile M=128, N=128, Kstep=32
//    Double-buffered: TDM (A) + vector loads (X) stage tile i+1 while the
//    WMMAs consume tile i; one barrier per K-step.
// ---------------------------------------------------------------------------
template <bool FINAL>
__global__ void gemm_kernel(const __bf16* __restrict__ A,   // [256,256] bf16
                            const __bf16* __restrict__ X,   // [B,256,6400] bf16
                            const float*  __restrict__ bias,
                            float bias_scale,
                            const float*  __restrict__ resid, // x, FINAL only
                            __bf16* __restrict__ outb,
                            float*  __restrict__ outf) {
  const int n0 = blockIdx.x * 128;
  const int m0 = blockIdx.y * 128;
  const int bb = blockIdx.z;
  const int tid  = threadIdx.x;
  const int wv   = tid >> 5;
  const int lane = tid & 31;
  const int hf   = lane >> 4;
  const int lr   = lane & 15;

  __shared__ __align__(16) __bf16 As[2][128 * 32];   // [m][k]
  __shared__ __align__(16) __bf16 Bs[2][128 * 32];   // [n][k] (transposed)

  const __bf16* Xb = X + (size_t)bb * C_ * HW_;

  // stage X tile [32 x 128] -> Bs[buf][n][k]; K-row pairs -> packed u32 stores
  auto stageB = [&](int k0, int buf) {
    int kp = tid >> 4;                 // 0..15 -> K rows 2kp, 2kp+1
    int n8 = tid & 15;                 // 8-col group
    uint4 r0 = *(const uint4*)(Xb + (size_t)(k0 + 2 * kp)     * HW_ + n0 + n8 * 8);
    uint4 r1 = *(const uint4*)(Xb + (size_t)(k0 + 2 * kp + 1) * HW_ + n0 + n8 * 8);
    const unsigned short* p0 = (const unsigned short*)&r0;
    const unsigned short* p1 = (const unsigned short*)&r1;
#pragma unroll
    for (int j = 0; j < 8; ++j) {
      unsigned val = (unsigned)p0[j] | ((unsigned)p1[j] << 16);
      *(unsigned*)(&Bs[buf][(n8 * 8 + j) * 32 + 2 * kp]) = val;
    }
  };
#if !defined(USE_TDM)
  auto stageA = [&](int k0, int buf) {
#pragma unroll
    for (int i = 0; i < 2; ++i) {
      int s   = tid * 2 + i;            // 0..511
      int row = s >> 2, c4 = s & 3;
      uint4 v = *(const uint4*)(A + (size_t)(m0 + row) * C_ + k0 + c4 * 8);
      *(uint4*)(&As[buf][row * 32 + c4 * 8]) = v;
    }
  };
#endif

  v8f acc[8];
#pragma unroll
  for (int t = 0; t < 8; ++t) acc[t] = vzero8();

  // prologue: stage tile 0
#if defined(USE_TDM)
  if (wv == 0) {
    tdm_issue_A(A + (size_t)m0 * C_, As[0]);
    __builtin_amdgcn_s_wait_tensorcnt(0);
  }
#else
  stageA(0, 0);
#endif
  stageB(0, 0);
  __syncthreads();

  for (int it = 0; it < 8; ++it) {
    const int k0  = it * 32;
    const int cur = it & 1;
    if (it < 7) {
#if defined(USE_TDM)
      if (wv == 0) tdm_issue_A(A + (size_t)m0 * C_ + k0 + 32, As[cur ^ 1]);
#else
      stageA(k0 + 32, cur ^ 1);
#endif
      stageB(k0 + 32, cur ^ 1);
    }
    if (k0 + 64 < C_)   // prefetch tile after next into GL2
      __builtin_prefetch(Xb + (size_t)(k0 + 64 + (tid >> 4)) * HW_ + n0 + (tid & 15) * 8, 0, 0);

    // A fragment (rows wv*16 .. +16)
    AF a;
#pragma unroll
    for (int v4 = 0; v4 < 8; ++v4) {
      int Kp = (v4 < 4) ? (2 * v4 + 8 * hf) : (2 * v4 + 8 + 8 * hf);
      a.u[v4] = *(const unsigned*)(&As[cur][(wv * 16 + lr) * 32 + Kp]);
    }
#pragma unroll
    for (int t = 0; t < 8; ++t) {
      AF bfr;
#pragma unroll
      for (int v4 = 0; v4 < 8; ++v4) {
        int Kp = (v4 < 4) ? (2 * v4 + 8 * hf) : (2 * v4 + 8 + 8 * hf);
        bfr.u[v4] = *(const unsigned*)(&Bs[cur][(t * 16 + lr) * 32 + Kp]);
      }
      acc[t] = wmma_bf16(a.v, bfr.v, acc[t]);
    }
#if defined(USE_TDM)
    if (wv == 0 && it < 7) __builtin_amdgcn_s_wait_tensorcnt(0);
#endif
    __syncthreads();
  }

  // epilogue
#pragma unroll
  for (int t = 0; t < 8; ++t) {
#pragma unroll
    for (int e = 0; e < 8; ++e) {
      int m = m0 + wv * 16 + e + 8 * hf;
      int n = n0 + t * 16 + lr;
      size_t addr = ((size_t)bb * C_ + m) * HW_ + n;
      float v = acc[t][e] + bias[m] * bias_scale;
      if (FINAL) outf[addr] = v + resid[addr];
      else       outb[addr] = (__bf16)v;
    }
  }
}

// ---------------------------------------------------------------------------
// 4. pack to_h / to_w layouts: P[(b*8+hd)][row][pix*32+c]
// ---------------------------------------------------------------------------
__global__ void pack_h_kernel(const __bf16* __restrict__ t, __bf16* __restrict__ P) {
  size_t g = (size_t)blockIdx.x * blockDim.x + threadIdx.x;
  if (g >= SZ_) return;
  int w = (int)(g % W_);
  int h = (int)((g / W_) % H_);
  int cc = (int)((g / HW_) % C_);
  int bb = (int)(g / ((size_t)C_ * HW_));
  int hd = cc / CP_, c = cc % CP_;
  P[(((size_t)(bb * HEADS_ + hd) * H_) + h) * D_ + w * CP_ + c] = t[g];
}

__global__ void pack_w_kernel(const __bf16* __restrict__ t, __bf16* __restrict__ P) {
  size_t g = (size_t)blockIdx.x * blockDim.x + threadIdx.x;
  if (g >= SZ_) return;
  int w = (int)(g % W_);
  int h = (int)((g / W_) % H_);
  int cc = (int)((g / HW_) % C_);
  int bb = (int)(g / ((size_t)C_ * HW_));
  int hd = cc / CP_, c = cc % CP_;
  P[(((size_t)(bb * HEADS_ + hd) * W_) + w) * D_ + h * CP_ + c] = t[g];
}

// ---------------------------------------------------------------------------
// 5. inverse L2 row norms over D=2560: dst[row] = 1/max(||.||, 1e-12)
// ---------------------------------------------------------------------------
__global__ void rownorm_kernel(const __bf16* __restrict__ src, float* __restrict__ dst) {
  int row = blockIdx.x;
  __shared__ float red[256];
  float s = 0.f;
  const __bf16* r = src + (size_t)row * D_;
  for (int i = threadIdx.x; i < D_; i += 256) {
    float v = (float)r[i];
    s += v * v;
  }
  red[threadIdx.x] = s;
  __syncthreads();
  for (int off = 128; off > 0; off >>= 1) {
    if (threadIdx.x < off) red[threadIdx.x] += red[threadIdx.x + off];
    __syncthreads();
  }
  if (threadIdx.x == 0) dst[row] = 1.0f / fmaxf(sqrtf(red[0]), 1e-12f);
}

// ---------------------------------------------------------------------------
// 6. axis attention per (b,head): S = Qn Kn^T, softmax, O = S*V + Qn, scatter
//    block = 160 threads (5 waves); wave wv handles rows [16wv, 16wv+16)
//    K tile (normalized) and V chunks double-buffered in LDS.
// ---------------------------------------------------------------------------
__global__ void attn_kernel(const __bf16* __restrict__ Q,   // packed [128][80][2560]
                            const __bf16* __restrict__ KV,  // packed (K and V)
                            const float*  __restrict__ rnq, // [128*80]
                            const float*  __restrict__ rnk,
                            __bf16* __restrict__ out,       // [B,C,H,W] layout
                            int axis)                       // 0: to_h, 1: to_w
{
  const int inst = blockIdx.x;         // b*8 + hd
  const int wv   = threadIdx.x >> 5;   // 0..4
  const int lane = threadIdx.x & 31;
  const int hf   = lane >> 4;
  const int lr   = lane & 15;
  const size_t base = (size_t)inst * H_ * D_;
  const int rb = inst * H_;            // row-norm base

  __shared__ __align__(16) __bf16 Ks[2][80 * 32];   // normalized K [col][k]
  __shared__ __align__(16) __bf16 Pl[5 * 16 * 96];  // softmaxed P (per wave)
  __shared__ __align__(16) __bf16 Vs[2][64 * 96];   // V chunk [n][k], K pad 96

  // stage normalized K tile: 80 cols x 32 K; contiguous packed b128 stores
  auto stageK = [&](int k0, int buf) {
#pragma unroll
    for (int i = 0; i < 2; ++i) {
      int seg = threadIdx.x * 2 + i;            // 0..319
      int col = seg >> 2, j4 = seg & 3;
      uint4 v = *(const uint4*)(KV + base + (size_t)col * D_ + k0 + j4 * 8);
      const __bf16* pv = (const __bf16*)&v;
      float sc = rnk[rb + col];
      Pk8 pk;
#pragma unroll
      for (int j = 0; j < 8; ++j) pk.h[j] = (__bf16)((float)pv[j] * sc);
      *(uint4*)(&Ks[buf][col * 32 + j4 * 8]) = pk.q;
    }
  };
  // stage V chunk: rows k=0..79 (pairs), cols nb..nb+64 -> Vs[buf][n][k]
  auto stageV = [&](int nb, int buf) {
#pragma unroll
    for (int i = 0; i < 2; ++i) {
      int seg = threadIdx.x * 2 + i;            // 0..319
      int kp = seg >> 3, c8 = seg & 7;          // K rows 2kp, 2kp+1
      uint4 r0 = *(const uint4*)(KV + base + (size_t)(2 * kp)     * D_ + nb + c8 * 8);
      uint4 r1 = *(const uint4*)(KV + base + (size_t)(2 * kp + 1) * D_ + nb + c8 * 8);
      const unsigned short* p0 = (const unsigned short*)&r0;
      const unsigned short* p1 = (const unsigned short*)&r1;
#pragma unroll
      for (int j = 0; j < 8; ++j) {
        unsigned val = (unsigned)p0[j] | ((unsigned)p1[j] << 16);
        *(unsigned*)(&Vs[buf][(c8 * 8 + j) * 96 + 2 * kp]) = val;
      }
    }
  };

  // ---- S = Qn * Kn^T  (per wave: M=16 rows, N=80, K=2560) ----
  v8f s[5];
#pragma unroll
  for (int t = 0; t < 5; ++t) s[t] = vzero8();

  stageK(0, 0);
  __syncthreads();
  for (int it = 0; it < 80; ++it) {
    const int k0  = it * 32;
    const int cur = it & 1;
    if (it + 1 < 80) stageK(k0 + 32, cur ^ 1);

    // A fragment: normalized Q rows of this wave, from global
    AF a;
    {
      int row = wv * 16 + lr;
      float sc = rnq[rb + row];
      const __bf16* qr = Q + base + (size_t)row * D_ + k0;
#pragma unroll
      for (int e = 0; e < 16; ++e)
        a.v[e] = (__bf16)((float)qr[amapK(e, hf)] * sc);
    }
#pragma unroll
    for (int t = 0; t < 5; ++t) {
      AF bfr;
#pragma unroll
      for (int v4 = 0; v4 < 8; ++v4) {
        int Kp = (v4 < 4) ? (2 * v4 + 8 * hf) : (2 * v4 + 8 + 8 * hf);
        bfr.u[v4] = *(const unsigned*)(&Ks[cur][(t * 16 + lr) * 32 + Kp]);
      }
      s[t] = wmma_bf16(a.v, bfr.v, s[t]);
    }
    __syncthreads();
  }

  // ---- softmax over 80 cols; row r = e + 8*hf lives in one 16-lane half ----
#pragma unroll
  for (int e = 0; e < 8; ++e) {
    float mx = -3.4e38f;
#pragma unroll
    for (int t = 0; t < 5; ++t) mx = fmaxf(mx, s[t][e]);
#pragma unroll
    for (int off = 1; off < 16; off <<= 1) mx = fmaxf(mx, __shfl_xor(mx, off, 32));
    float sum = 0.f;
#pragma unroll
    for (int t = 0; t < 5; ++t) {
      float ex = __expf(s[t][e] - mx);
      s[t][e] = ex;
      sum += ex;
    }
#pragma unroll
    for (int off = 1; off < 16; off <<= 1) sum += __shfl_xor(sum, off, 32);
    float inv = 1.0f / sum;
#pragma unroll
    for (int t = 0; t < 5; ++t) s[t][e] *= inv;
  }

  // stash P (bf16) into per-wave LDS region, K padded 80->96 with zeros
#pragma unroll
  for (int t = 0; t < 5; ++t)
#pragma unroll
    for (int e = 0; e < 8; ++e)
      Pl[(wv * 16 + e + 8 * hf) * 96 + t * 16 + lr] = (__bf16)s[t][e];
  if (lane < 16) {
#pragma unroll
    for (int kp = 80; kp < 96; ++kp) Pl[(wv * 16 + lane) * 96 + kp] = (__bf16)0.0f;
  }
  // zero the K-pad region of both V buffers (rows 80..95 never restaged)
  for (int idx = threadIdx.x; idx < 2 * 64 * 16; idx += 160) {
    int buf = idx >> 10;
    int r   = idx & 1023;
    int n = r >> 4, kp = 80 + (r & 15);
    Vs[buf][n * 96 + kp] = (__bf16)0.0f;
  }

  // ---- O = P * V  (M=16, N=2560 in chunks of 64, K=80 padded to 96) ----
  const int bb = inst >> 3, hd = inst & 7;
  stageV(0, 0);
  __syncthreads();
  for (int nc = 0; nc < 40; ++nc) {
    const int nb  = nc * 64;
    const int cur = nc & 1;
    if (nc + 1 < 40) stageV(nb + 64, cur ^ 1);

    v8f o[4];
#pragma unroll
    for (int t = 0; t < 4; ++t) o[t] = vzero8();
#pragma unroll
    for (int ks = 0; ks < 3; ++ks) {
      const int k0 = ks * 32;
      AF a;
#pragma unroll
      for (int v4 = 0; v4 < 8; ++v4) {
        int Kp = (v4 < 4) ? (2 * v4 + 8 * hf) : (2 * v4 + 8 + 8 * hf);
        a.u[v4] = *(const unsigned*)(&Pl[(wv * 16 + lr) * 96 + k0 + Kp]);
      }
#pragma unroll
      for (int t = 0; t < 4; ++t) {
        AF bfr;
#pragma unroll
        for (int v4 = 0; v4 < 8; ++v4) {
          int Kp = (v4 < 4) ? (2 * v4 + 8 * hf) : (2 * v4 + 8 + 8 * hf);
          bfr.u[v4] = *(const unsigned*)(&Vs[cur][(t * 16 + lr) * 96 + k0 + Kp]);
        }
        o[t] = wmma_bf16(a.v, bfr.v, o[t]);
      }
    }
    // epilogue: + Qn residual, scatter via from_h / from_w
#pragma unroll
    for (int t = 0; t < 4; ++t) {
#pragma unroll
      for (int e = 0; e < 8; ++e) {
        int row = wv * 16 + e + 8 * hf;       // sequence index (h or w)
        int col = nb + t * 16 + lr;           // pix*32 + c
        float resv = rnq[rb + row] * (float)Q[base + (size_t)row * D_ + col];
        float val = o[t][e] + resv;
        int pix = col >> 5, c = col & 31;
        size_t addr;
        if (axis == 0)
          addr = (((size_t)bb * C_ + hd * CP_ + c) * H_ + row) * W_ + pix;
        else
          addr = (((size_t)bb * C_ + hd * CP_ + c) * H_ + pix) * W_ + row;
        out[addr] = (__bf16)val;
      }
    }
    __syncthreads();
  }
}

// ---------------------------------------------------------------------------
// 7. ssum = outx + outy (bf16)
// ---------------------------------------------------------------------------
__global__ void add_kernel(const __bf16* __restrict__ a, const __bf16* __restrict__ b,
                           __bf16* __restrict__ o) {
  size_t g = (size_t)blockIdx.x * blockDim.x + threadIdx.x;
  if (g < SZ_) o[g] = (__bf16)((float)a[g] + (float)b[g]);
}

// ---------------------------------------------------------------------------
// launch
// ---------------------------------------------------------------------------
extern "C" void kernel_launch(void* const* d_in, const int* in_sizes, int n_in,
                              void* d_out, int out_size, void* d_ws, size_t ws_size,
                              hipStream_t stream) {
  const float* x    = (const float*)d_in[0];
  const float* ln_g = (const float*)d_in[1];
  const float* ln_b = (const float*)d_in[2];
  const float* w_in = (const float*)d_in[3];
  const float* b_in = (const float*)d_in[4];
  const float* w_out= (const float*)d_in[5];
  const float* b_out= (const float*)d_in[6];
  const float* wx7  = (const float*)d_in[7];
  const float* bx7  = (const float*)d_in[8];
  const float* wx11 = (const float*)d_in[9];
  const float* bx11 = (const float*)d_in[10];
  const float* wx21 = (const float*)d_in[11];
  const float* bx21 = (const float*)d_in[12];
  const float* wy7  = (const float*)d_in[13];
  const float* by7  = (const float*)d_in[14];
  const float* wy11 = (const float*)d_in[15];
  const float* by11 = (const float*)d_in[16];
  const float* wy21 = (const float*)d_in[17];
  const float* by21 = (const float*)d_in[18];
  float* outp = (float*)d_out;

  char* ws = (char*)d_ws;
  const size_t BUF = SZ_ * 2;              // one bf16 [B,C,H,W] buffer
  __bf16* xn    = (__bf16*)(ws + 0 * BUF);
  __bf16* sx    = (__bf16*)(ws + 1 * BUF); // later: outx
  __bf16* sy    = (__bf16*)(ws + 2 * BUF); // later: outy
  __bf16* fx    = (__bf16*)(ws + 3 * BUF); // later: ssum
  __bf16* fy    = (__bf16*)(ws + 4 * BUF);
  __bf16* ph_fx = (__bf16*)(ws + 5 * BUF);
  __bf16* ph_fy = (__bf16*)(ws + 6 * BUF);
  __bf16* pw_fx = (__bf16*)(ws + 7 * BUF);
  __bf16* pw_fy = (__bf16*)(ws + 8 * BUF);
  char* tail = ws + 9 * BUF;
  float* n_qh = (float*)(tail);                     // 128*80
  float* n_kh = (float*)(tail + 1 * 40960);
  float* n_qw = (float*)(tail + 2 * 40960);
  float* n_kw = (float*)(tail + 3 * 40960);
  __bf16* w_in_bf  = (__bf16*)(tail + 4 * 40960);
  __bf16* w_out_bf = (__bf16*)(tail + 4 * 40960 + C_ * C_ * 2);
  __bf16* outx = sx;
  __bf16* outy = sy;
  __bf16* ssum = fx;

  // 0. weights -> bf16
  prep_weights_kernel<<<(2 * C_ * C_ + 255) / 256, 256, 0, stream>>>(
      w_in, w_out, w_in_bf, w_out_bf);
  // 1. LayerNorm
  ln_kernel<<<NPIX_ / 256, 256, 0, stream>>>(x, ln_g, ln_b, xn);
  // 2. depthwise sums
  dwx_kernel<<<B_ * C_ * H_, 128, 0, stream>>>(xn, wx7, bx7, wx11, bx11, wx21, bx21, sx);
  dwy_kernel<<<dim3(B_ * C_, H_ / 16), 256, 0, stream>>>(xn, wy7, by7, wy11, by11, wy21, by21, sy);
  // 3. pointwise GEMMs (w_in)
  dim3 ggrid(HW_ / 128, C_ / 128, B_);
  gemm_kernel<false><<<ggrid, 256, 0, stream>>>(w_in_bf, sx, b_in, 1.0f, nullptr, fx, nullptr);
  gemm_kernel<false><<<ggrid, 256, 0, stream>>>(w_in_bf, sy, b_in, 1.0f, nullptr, fy, nullptr);
  // 4. pack attention layouts
  int pgrid = (int)(SZ_ / 256);
  pack_h_kernel<<<pgrid, 256, 0, stream>>>(fx, ph_fx);
  pack_h_kernel<<<pgrid, 256, 0, stream>>>(fy, ph_fy);
  pack_w_kernel<<<pgrid, 256, 0, stream>>>(fx, pw_fx);
  pack_w_kernel<<<pgrid, 256, 0, stream>>>(fy, pw_fy);
  // 5. row norms (q=fy_h, k=fx_h for axis h; q=fx_w, k=fy_w for axis w)
  rownorm_kernel<<<B_ * HEADS_ * H_, 256, 0, stream>>>(ph_fy, n_qh);
  rownorm_kernel<<<B_ * HEADS_ * H_, 256, 0, stream>>>(ph_fx, n_kh);
  rownorm_kernel<<<B_ * HEADS_ * W_, 256, 0, stream>>>(pw_fx, n_qw);
  rownorm_kernel<<<B_ * HEADS_ * W_, 256, 0, stream>>>(pw_fy, n_kw);
  // 6. attention (axis h then axis w)
  attn_kernel<<<B_ * HEADS_, 160, 0, stream>>>(ph_fy, ph_fx, n_qh, n_kh, outx, 0);
  attn_kernel<<<B_ * HEADS_, 160, 0, stream>>>(pw_fx, pw_fy, n_qw, n_kw, outy, 1);
  // 7. out_x + out_y
  add_kernel<<<pgrid, 256, 0, stream>>>(outx, outy, ssum);
  // 8. final GEMM (w_out) + 2*b_out + x -> d_out fp32
  gemm_kernel<true><<<ggrid, 256, 0, stream>>>(w_out_bf, ssum, b_out, 2.0f, x, nullptr, outp);
}
